// GraphTransformerInvariant_58171037057262
// MI455X (gfx1250) — compile-verified, matrix operand order checked by reading, not run on hardware
//
#include <hip/hip_runtime.h>
#include <math.h>

typedef __bf16 bf16;
typedef __attribute__((ext_vector_type(16))) __bf16 v16bf;
typedef __attribute__((ext_vector_type(8)))  float  v8f;

static constexpr int Bsz = 32, NMAX = 256, Lnum = 4, Dim = 256, Hn = 8, Qn = 4, Pn = 16, Fn = 32;
static constexpr int Ntot = Bsz * NMAX;   // 8192
static constexpr int DH   = Dim / Hn;     // 32
static constexpr int QP   = Qn * Pn;      // 64

// dynamic LDS layout for attention kernel (bytes)
static constexpr int SMEM_K   = 0;                          // 256*32 bf16
static constexpr int SMEM_VT  = SMEM_K  + 256 * 32 * 2;     // 32*256 bf16
static constexpr int SMEM_ZR  = SMEM_VT + 32 * 256 * 2;     // 256*64 bf16
static constexpr int SMEM_ZI  = SMEM_ZR + 256 * 64 * 2;     // 256*64 bf16
static constexpr int SMEM_P   = SMEM_ZI + 256 * 64 * 2;     // 8 waves * 16*32 bf16
static constexpr int SMEM_PHI = SMEM_P  + 8 * 16 * 32 * 2;  // 64 f32
static constexpr int SMEM_SZ  = SMEM_PHI + QP * 4;

// ---------------- CDNA5 async copy helpers ----------------
// GVS addressing: mem = SGPR base + VGPR offset + inst offset; data lands in LDS at
// the byte address in the "vdst" VGPR (generic LDS pointers carry the LDS offset in
// their low 32 bits). Tracked by ASYNCcnt.
__device__ __forceinline__ void async_load_b128(unsigned lds_addr, unsigned goff,
                                                const void* base) {
  asm volatile("global_load_async_to_lds_b128 %0, %1, %2 offset:0"
               :: "v"(lds_addr), "v"(goff), "s"(base)
               : "memory");
}
__device__ __forceinline__ void wait_async0() {
  asm volatile("s_wait_asynccnt 0x0" ::: "memory");
}
__device__ __forceinline__ void wait_async3() {
  asm volatile("s_wait_asynccnt 0x3" ::: "memory");
}

// ---------------- WMMA fragment helpers (16x16x32 bf16) ----------------
// A fragment (16xK=32): lane L holds row M=L&15; element e -> K = (e&7) + 16*(e>>3) + 8*(L>>4)
// i.e. two contiguous 8-element runs at K = 8*(L>>4) and K = 16 + 8*(L>>4).
// B fragment loads identically from B^T stored row-major [N][K].
__device__ __forceinline__ v16bf load_frag(const bf16* p, int ld, int rb, int kb, int lane) {
  v16bf f;
  const bf16* q = p + (rb + (lane & 15)) * ld + kb + ((lane >> 4) << 3);
#pragma unroll
  for (int e = 0; e < 8; ++e) { f[e] = q[e]; f[8 + e] = q[16 + e]; }
  return f;
}

__device__ __forceinline__ v16bf load_frag_scaled(const bf16* p, int ld, int rb, int kb,
                                                  int lane, const float* s) {
  v16bf f;
  const int half8 = (lane >> 4) << 3;
  const bf16* q = p + (rb + (lane & 15)) * ld + kb + half8;
  const float* sq = s + kb + half8;
#pragma unroll
  for (int e = 0; e < 8; ++e) {
    f[e]     = (bf16)((float)q[e]      * sq[e]);
    f[8 + e] = (bf16)((float)q[16 + e] * sq[16 + e]);
  }
  return f;
}

__device__ __forceinline__ v8f wmma_bf16(v16bf a, v16bf b, v8f c) {
  return __builtin_amdgcn_wmma_f32_16x16x32_bf16(false, a, false, b, (short)0, c, false, false);
}

// ---------------- prep kernels ----------------
__global__ void wprep_kernel(const float* __restrict__ W, bf16* __restrict__ Wt) {
  // Wt[(l*D + dout)*D + din] = W[(l*D + din)*D + dout]
  int idx = blockIdx.x * blockDim.x + threadIdx.x;     // L*D*D total
  int l = idx >> 16;
  int rem = idx & 65535;
  int n = rem >> 8, k = rem & 255;
  Wt[idx] = (bf16)W[(l * Dim + k) * Dim + n];
}

__global__ void zprep_kernel(const float* __restrict__ z, bf16* __restrict__ zbr,
                             bf16* __restrict__ zbi) {
  int idx = blockIdx.x * blockDim.x + threadIdx.x;     // Ntot*QP total
  int node = idx >> 6, qp = idx & 63;
  zbr[idx] = (bf16)z[(node << 7) + qp * 2];
  zbi[idx] = (bf16)z[(node << 7) + qp * 2 + 1];
}

__global__ void f2b_kernel(const float* __restrict__ src, bf16* __restrict__ dst, int n) {
  int i = blockIdx.x * blockDim.x + threadIdx.x;
  if (i < n) dst[i] = (bf16)src[i];
}

// phi(Lambda): tiny MLP, phibuf laid out [b][h][qp]
__global__ void phi_kernel(const float* __restrict__ Lambda, const float* __restrict__ W1,
                           const float* __restrict__ b1, const float* __restrict__ W2,
                           const float* __restrict__ b2, float* __restrict__ phibuf, int layer) {
  int id = blockIdx.x * blockDim.x + threadIdx.x;      // Bsz*QP = 2048
  if (id >= Bsz * QP) return;
  int b = id / QP, qp = id % QP;
  float lam = Lambda[b * QP + qp];
  const float* w1  = W1 + layer * Fn;
  const float* bb1 = b1 + layer * Fn;
  const float* w2  = W2 + layer * Fn * Hn;
  const float* bb2 = b2 + layer * Hn;
  float hf[Fn];
#pragma unroll
  for (int f = 0; f < Fn; ++f) {
    float v = lam * w1[f] + bb1[f];
    hf[f] = v > 0.f ? v : 0.f;
  }
#pragma unroll
  for (int h = 0; h < Hn; ++h) {
    float o = bb2[h];
#pragma unroll
    for (int f = 0; f < Fn; ++f) o += hf[f] * w2[f * Hn + h];
    phibuf[(b * Hn + h) * QP + qp] = o;
  }
}

// ---------------- tiled WMMA GEMM: C[M,N] = A[M,K] * Bt[N,K]^T ----------------
// Double-buffered LDS staging via async-copy engine (ASYNCcnt pipeline).
// MODE 0: out_bf16 = (acc + bias) * scale         (QKV projections)
// MODE 1: out_f32  = acc + bias + resid           (output projection + residual)
template <int MODE>
__global__ __launch_bounds__(256) void gemm_kernel(
    const bf16* __restrict__ A, const bf16* __restrict__ Bt,
    const float* __restrict__ bias, const float* __restrict__ resid,
    bf16* __restrict__ outb, float* __restrict__ outf,
    int M, int N, int K, float scaleOut) {
  __shared__ bf16 sA[2][64 * 32];
  __shared__ bf16 sB[2][128 * 32];
  const int tid = threadIdx.x, lane = tid & 31, wave = tid >> 5;
  const int m0 = blockIdx.x * 64, n0 = blockIdx.y * 128;
  const int wm = (wave & 1) * 32, wn = (wave >> 1) * 32;

  // per-thread async-copy slots: A -> one 16B chunk, Bt -> two 16B chunks
  const int arow = tid >> 2, acol = (tid & 3) * 8;   // 64x32: 8 bf16 per thread
  const int brow = tid >> 1, bcol = (tid & 1) * 16;  // 128x32: 16 bf16 per thread

  auto issue = [&](int kc, int buf) {
    unsigned la = (unsigned)(uintptr_t)&sA[buf][arow * 32 + acol];
    unsigned ga = (unsigned)(((m0 + arow) * K + kc + acol) * 2);
    async_load_b128(la, ga, A);
    unsigned lb = (unsigned)(uintptr_t)&sB[buf][brow * 32 + bcol];
    unsigned gb = (unsigned)(((n0 + brow) * K + kc + bcol) * 2);
    async_load_b128(lb, gb, Bt);
    async_load_b128(lb + 16, gb + 16, Bt);
  };

  v8f acc[2][2] = {};
  const int nk = K / 32;
  issue(0, 0);
  for (int ki = 0; ki < nk; ++ki) {
    const int buf = ki & 1;
    __syncthreads();  // everyone done computing from buf^1 before overwriting it
    const int knext = (ki + 1 < nk) ? (ki + 1) * 32 : 0;  // last issue is a harmless wrap
    issue(knext, buf ^ 1);
    wait_async3();    // this wave's chunk-ki loads (3) complete; next chunk in flight
    __syncthreads();  // all waves' chunk-ki data visible in LDS
    v16bf a0 = load_frag(sA[buf], 32, wm, 0, lane);
    v16bf a1 = load_frag(sA[buf], 32, wm + 16, 0, lane);
    v16bf b0 = load_frag(sB[buf], 32, wn, 0, lane);
    v16bf b1 = load_frag(sB[buf], 32, wn + 16, 0, lane);
    acc[0][0] = wmma_bf16(a0, b0, acc[0][0]);
    acc[0][1] = wmma_bf16(a0, b1, acc[0][1]);
    acc[1][0] = wmma_bf16(a1, b0, acc[1][0]);
    acc[1][1] = wmma_bf16(a1, b1, acc[1][1]);
  }
#pragma unroll
  for (int mi = 0; mi < 2; ++mi)
#pragma unroll
    for (int ni = 0; ni < 2; ++ni)
#pragma unroll
      for (int r = 0; r < 8; ++r) {
        int row = m0 + wm + mi * 16 + r + ((lane >> 4) << 3);
        int col = n0 + wn + ni * 16 + (lane & 15);
        float v = acc[mi][ni][r] + bias[col];
        if (MODE == 0) {
          outb[(size_t)row * N + col] = (bf16)(v * scaleOut);
        } else {
          outf[(size_t)row * N + col] = v + resid[(size_t)row * N + col];
        }
      }
}

// ---------------- fused attention (one workgroup per (b,h)) ----------------
// scores = q·k^T (q pre-scaled by 1/sqrt(dh)) + (phi*zr)·zr^T + (phi*zi)·zi^T; softmax; P·V
// All operands staged in (dynamic) LDS so the score loop is pure ds_load -> v_wmma.
__global__ __launch_bounds__(256) void attn_kernel(
    const bf16* __restrict__ Qb, const bf16* __restrict__ Kb, const bf16* __restrict__ Vb,
    const bf16* __restrict__ zbr, const bf16* __restrict__ zbi,
    const float* __restrict__ phibuf, bf16* __restrict__ Ob) {
  extern __shared__ char smem[];
  bf16*  sK   = (bf16*)(smem + SMEM_K);    // [j][d]  256x32  (= B^T for q.k^T)
  bf16*  sVt  = (bf16*)(smem + SMEM_VT);   // [d][j]  32x256  (= B^T for P.V)
  bf16*  sZr  = (bf16*)(smem + SMEM_ZR);   // [j][qp] 256x64
  bf16*  sZi  = (bf16*)(smem + SMEM_ZI);   // [j][qp] 256x64
  bf16*  sP   = (bf16*)(smem + SMEM_P);    // per-wave 16x32 prob staging
  float* sphi = (float*)(smem + SMEM_PHI); // 64

  const int b = blockIdx.x >> 3, h = blockIdx.x & 7;
  const int tid = threadIdx.x, lane = tid & 31, wave = tid >> 5;
  const int nodeBase = b * NMAX;
  const int hoff = h * DH;

  // Async-copy: each thread owns row j = tid of sK (64B) and sZr/sZi (128B each)
  {
    unsigned lk = (unsigned)(uintptr_t)&sK[tid * 32];
    unsigned gk = (unsigned)((((unsigned)(nodeBase + tid)) * Dim + hoff) * 2);
#pragma unroll
    for (int c = 0; c < 4; ++c) async_load_b128(lk + c * 16, gk + c * 16, Kb);
    unsigned lzr = (unsigned)(uintptr_t)&sZr[tid * QP];
    unsigned lzi = (unsigned)(uintptr_t)&sZi[tid * QP];
    unsigned gz  = (unsigned)(((unsigned)(nodeBase + tid)) * QP * 2);
#pragma unroll
    for (int c = 0; c < 8; ++c) {
      async_load_b128(lzr + c * 16, gz + c * 16, zbr);
      async_load_b128(lzi + c * 16, gz + c * 16, zbi);
    }
  }
  // V^T needs an element transpose -> regular loads
  for (int idx = tid; idx < 256 * 32; idx += 256) {
    int j = idx >> 5, d = idx & 31;
    sVt[d * 256 + j] = Vb[(size_t)(nodeBase + j) * Dim + hoff + d];
  }
  if (tid < QP) sphi[tid] = phibuf[(b * Hn + h) * QP + tid];
  wait_async0();
  __syncthreads();

  for (int s = 0; s < 2; ++s) {
    const int mrow = wave * 32 + s * 16;
    // A-side fragments (reused across all 16 n-tiles)
    v16bf qf   = load_frag(Qb + (size_t)nodeBase * Dim + hoff, Dim, mrow, 0, lane);
    v16bf zrw0 = load_frag_scaled(sZr, QP, mrow, 0,  lane, sphi);
    v16bf zrw1 = load_frag_scaled(sZr, QP, mrow, 32, lane, sphi);
    v16bf ziw0 = load_frag_scaled(sZi, QP, mrow, 0,  lane, sphi);
    v16bf ziw1 = load_frag_scaled(sZi, QP, mrow, 32, lane, sphi);

    v8f acc[16] = {};
#pragma unroll
    for (int t = 0; t < 16; ++t) {
      int nb = t * 16;
      v16bf kf  = load_frag(sK,  32, nb, 0,  lane);
      v16bf br0 = load_frag(sZr, QP, nb, 0,  lane);
      v16bf br1 = load_frag(sZr, QP, nb, 32, lane);
      v16bf bi0 = load_frag(sZi, QP, nb, 0,  lane);
      v16bf bi1 = load_frag(sZi, QP, nb, 32, lane);
      v8f c = acc[t];
      c = wmma_bf16(zrw0, br0, c);
      c = wmma_bf16(zrw1, br1, c);
      c = wmma_bf16(ziw0, bi0, c);
      c = wmma_bf16(ziw1, bi1, c);
      c = wmma_bf16(qf, kf, c);
      acc[t] = c;
    }

    // softmax per row (rows live across 16-lane halves; xor masks 1..8 stay in-half)
    float inv[8];
#pragma unroll
    for (int r = 0; r < 8; ++r) {
      float m = -1e30f;
#pragma unroll
      for (int t = 0; t < 16; ++t) m = fmaxf(m, acc[t][r]);
      m = fmaxf(m, __shfl_xor(m, 1));
      m = fmaxf(m, __shfl_xor(m, 2));
      m = fmaxf(m, __shfl_xor(m, 4));
      m = fmaxf(m, __shfl_xor(m, 8));
      float ssum = 0.f;
#pragma unroll
      for (int t = 0; t < 16; ++t) {
        float e = __expf(acc[t][r] - m);
        acc[t][r] = e;
        ssum += e;
      }
      ssum += __shfl_xor(ssum, 1);
      ssum += __shfl_xor(ssum, 2);
      ssum += __shfl_xor(ssum, 4);
      ssum += __shfl_xor(ssum, 8);
      inv[r] = 1.f / ssum;
    }

    // O = P . V   (stage probs through per-wave LDS to get A-fragment layout)
    bf16* sPw = sP + wave * (16 * 32);
    v8f o0 = {}, o1 = {};
    for (int kc = 0; kc < 8; ++kc) {
#pragma unroll
      for (int tt = 0; tt < 2; ++tt) {
        int t = kc * 2 + tt;
#pragma unroll
        for (int r = 0; r < 8; ++r) {
          int rr = r + ((lane >> 4) << 3);
          sPw[rr * 32 + tt * 16 + (lane & 15)] = (bf16)(acc[t][r] * inv[r]);
        }
      }
      __builtin_amdgcn_wave_barrier();
      asm volatile("s_wait_dscnt 0" ::: "memory");
      v16bf pf  = load_frag(sPw, 32, 0, 0, lane);
      v16bf vb0 = load_frag(sVt, 256, 0,  kc * 32, lane);
      v16bf vb1 = load_frag(sVt, 256, 16, kc * 32, lane);
      o0 = wmma_bf16(pf, vb0, o0);
      o1 = wmma_bf16(pf, vb1, o1);
      __builtin_amdgcn_wave_barrier();
      asm volatile("s_wait_dscnt 0" ::: "memory");
    }
#pragma unroll
    for (int r = 0; r < 8; ++r) {
      int row = nodeBase + mrow + r + ((lane >> 4) << 3);
      Ob[(size_t)row * Dim + hoff + (lane & 15)]      = (bf16)o0[r];
      Ob[(size_t)row * Dim + hoff + 16 + (lane & 15)] = (bf16)o1[r];
    }
  }
}

// ---------------- batchnorm ----------------
__global__ __launch_bounds__(256) void bnstats_kernel(const float* __restrict__ hp,
                                                      float* __restrict__ mv) {
  __shared__ float s1[256], s2[256];
  int c = blockIdx.x, tid = threadIdx.x;
  float a = 0.f, b2 = 0.f;
  for (int r = tid; r < Ntot; r += 256) {
    float v = hp[(size_t)r * Dim + c];
    a += v; b2 += v * v;
  }
  s1[tid] = a; s2[tid] = b2;
  __syncthreads();
  for (int off = 128; off > 0; off >>= 1) {
    if (tid < off) { s1[tid] += s1[tid + off]; s2[tid] += s2[tid + off]; }
    __syncthreads();
  }
  if (tid == 0) {
    float mean = s1[0] / (float)Ntot;
    float var  = s2[0] / (float)Ntot - mean * mean;
    mv[c] = mean; mv[256 + c] = var;
  }
}

__global__ void bnapply_kernel(const float* __restrict__ hp, const float* __restrict__ mv,
                               const float* __restrict__ gamma, const float* __restrict__ beta,
                               float* __restrict__ h0out, bf16* __restrict__ hbout) {
  int idx = blockIdx.x * blockDim.x + threadIdx.x;
  int c = idx & 255;
  float mean = mv[c], var = mv[256 + c];
  float v = (hp[idx] - mean) * rsqrtf(var + 1e-5f) * gamma[c] + beta[c];
  h0out[idx] = v;
  hbout[idx] = (bf16)v;
}

// ---------------- host launcher ----------------
extern "C" void kernel_launch(void* const* d_in, const int* in_sizes, int n_in,
                              void* d_out, int out_size, void* d_ws, size_t ws_size,
                              hipStream_t stream) {
  const float* x      = (const float*)d_in[0];
  const float* z      = (const float*)d_in[1];
  const float* Lambda = (const float*)d_in[2];
  const float* Wq = (const float*)d_in[4];
  const float* Wk = (const float*)d_in[5];
  const float* Wv = (const float*)d_in[6];
  const float* Wo = (const float*)d_in[7];
  const float* bq = (const float*)d_in[8];
  const float* bk = (const float*)d_in[9];
  const float* bv = (const float*)d_in[10];
  const float* bo = (const float*)d_in[11];
  const float* phiW1 = (const float*)d_in[12];
  const float* phib1 = (const float*)d_in[13];
  const float* phiW2 = (const float*)d_in[14];
  const float* phib2 = (const float*)d_in[15];
  const float* gamma = (const float*)d_in[16];
  const float* beta  = (const float*)d_in[17];

  char* w = (char*)d_ws;
  size_t off = 0;
  auto take = [&](size_t bytes) {
    void* p = w + off;
    off += (bytes + 255) & ~(size_t)255;
    return p;
  };
  bf16* Wtq = (bf16*)take(sizeof(bf16) * Lnum * Dim * Dim);
  bf16* Wtk = (bf16*)take(sizeof(bf16) * Lnum * Dim * Dim);
  bf16* Wtv = (bf16*)take(sizeof(bf16) * Lnum * Dim * Dim);
  bf16* Wto = (bf16*)take(sizeof(bf16) * Lnum * Dim * Dim);
  bf16* zbr = (bf16*)take(sizeof(bf16) * Ntot * QP);
  bf16* zbi = (bf16*)take(sizeof(bf16) * Ntot * QP);
  float* phibuf = (float*)take(sizeof(float) * Bsz * Hn * QP);
  bf16* hb = (bf16*)take(sizeof(bf16) * Ntot * Dim);
  bf16* Qb = (bf16*)take(sizeof(bf16) * Ntot * Dim);
  bf16* Kb = (bf16*)take(sizeof(bf16) * Ntot * Dim);
  bf16* Vb = (bf16*)take(sizeof(bf16) * Ntot * Dim);
  bf16* Ob = (bf16*)take(sizeof(bf16) * Ntot * Dim);
  float* hpre  = (float*)take(sizeof(float) * Ntot * Dim);
  float* h0buf = (float*)take(sizeof(float) * Ntot * Dim);
  float* mv    = (float*)take(sizeof(float) * 512);

  wprep_kernel<<<Lnum * Dim * Dim / 256, 256, 0, stream>>>(Wq, Wtq);
  wprep_kernel<<<Lnum * Dim * Dim / 256, 256, 0, stream>>>(Wk, Wtk);
  wprep_kernel<<<Lnum * Dim * Dim / 256, 256, 0, stream>>>(Wv, Wtv);
  wprep_kernel<<<Lnum * Dim * Dim / 256, 256, 0, stream>>>(Wo, Wto);
  zprep_kernel<<<Ntot * QP / 256, 256, 0, stream>>>(z, zbr, zbi);
  f2b_kernel<<<Ntot * Dim / 256, 256, 0, stream>>>(x, hb, Ntot * Dim);

  const float* h0cur = x;
  const float qscale = 0.17677669529663687f;  // 1/sqrt(32)
  dim3 gg(Ntot / 64, Dim / 128);

  for (int i = 0; i < Lnum; ++i) {
    phi_kernel<<<8, 256, 0, stream>>>(Lambda, phiW1, phib1, phiW2, phib2, phibuf, i);
    gemm_kernel<0><<<gg, 256, 0, stream>>>(hb, Wtq + i * Dim * Dim, bq + i * Dim, nullptr,
                                           Qb, nullptr, Ntot, Dim, Dim, qscale);
    gemm_kernel<0><<<gg, 256, 0, stream>>>(hb, Wtk + i * Dim * Dim, bk + i * Dim, nullptr,
                                           Kb, nullptr, Ntot, Dim, Dim, 1.0f);
    gemm_kernel<0><<<gg, 256, 0, stream>>>(hb, Wtv + i * Dim * Dim, bv + i * Dim, nullptr,
                                           Vb, nullptr, Ntot, Dim, Dim, 1.0f);
    attn_kernel<<<Bsz * Hn, 256, SMEM_SZ, stream>>>(Qb, Kb, Vb, zbr, zbi, phibuf, Ob);
    gemm_kernel<1><<<gg, 256, 0, stream>>>(Ob, Wto + i * Dim * Dim, bo + i * Dim, h0cur,
                                           nullptr, hpre, Ntot, Dim, Dim, 1.0f);
    bnstats_kernel<<<Dim, 256, 0, stream>>>(hpre, mv);
    float* h0next = (i == Lnum - 1) ? (float*)d_out : h0buf;
    bnapply_kernel<<<Ntot * Dim / 256, 256, 0, stream>>>(hpre, mv, gamma + i * Dim,
                                                         beta + i * Dim, h0next, hb);
    h0cur = h0next;
  }

  // output tuple part 2: z passthrough
  hipMemcpyAsync((float*)d_out + (size_t)Ntot * Dim, z,
                 sizeof(float) * Ntot * QP * 2, hipMemcpyDeviceToDevice, stream);
}
